// mpnn_9139690405991
// MI455X (gfx1250) — matrix operand order repaired and checked by reading, not compiled, not run on hardware
//
#include <hip/hip_runtime.h>

typedef __bf16 bf16;
typedef __attribute__((ext_vector_type(16))) __bf16 bf16x16;
typedef __attribute__((ext_vector_type(8)))  __bf16 bf16x8;
typedef __attribute__((ext_vector_type(4)))  __bf16 bf16x4;
typedef __attribute__((ext_vector_type(8)))  float  f32x8;
typedef __attribute__((ext_vector_type(4)))  float  f32x4;

#define NAGT 1024   // NA == NT == 1024
#define DIM  128    // A_DIM == E_DIM == K == 128
#define LLAB 128    // label count

// ---------------------------------------------------------------------------
// Projection: C[row, l, n] = sum_k U[row, k] * Wt[l, n, k]
//   U  : [1024, 128] bf16 (row-major state)
//   Wt : [L, 128(N), 128(K)] bf16 (weights pre-transposed so K is contiguous)
//   C  : [1024, L, 128] f32
// One wave computes one 16x16 tile for one label; block = 8 waves covering N.
// ---------------------------------------------------------------------------
__global__ __launch_bounds__(256)
void proj_wmma(const bf16* __restrict__ U,
               const bf16* __restrict__ Wt,
               float* __restrict__ C)
{
    const int lane  = threadIdx.x & 31;
    const int wave  = threadIdx.x >> 5;     // 0..7 -> 16-wide N tile
    const int mBase = blockIdx.x * 16;      // row tile
    const int l     = blockIdx.y;           // label
    const int nBase = wave * 16;

    const int half = lane >> 4;             // 0: lanes 0-15, 1: lanes 16-31
    const int l16  = lane & 15;

    // A operand: row = mBase + (lane%16); elements 0..15 hold
    // K = kk + half*16 + 0..15 (same order as B)
    const bf16* Arow = U + (size_t)(mBase + l16) * DIM;
    // B operand: lane owns column nBase+l16; elements 0..15 hold
    // K = kk + half*16 + 0..15 (contiguous thanks to pre-transposed weights)
    const bf16* Brow = Wt + ((size_t)l * DIM + (nBase + l16)) * DIM;

    f32x8 c = {};
#pragma unroll
    for (int kk = 0; kk < DIM; kk += 32) {
        bf16x8 alo = *(const bf16x8*)(Arow + kk + half * 16);
        bf16x8 ahi = *(const bf16x8*)(Arow + kk + half * 16 + 8);
        bf16x16 a;
#pragma unroll
        for (int i = 0; i < 8; ++i) { a[i] = alo[i]; a[i + 8] = ahi[i]; }

        bf16x16 b = *(const bf16x16*)(Brow + kk + half * 16);

        c = __builtin_amdgcn_wmma_f32_16x16x32_bf16(
                /*neg_a=*/false, a, /*neg_b=*/false, b,
                /*c_mod=*/(short)0, c, /*reuse_a=*/false, /*reuse_b=*/false);
    }

    // C/D layout: VGPR r -> row mBase + r + 8*half, col nBase + (lane%16)
#pragma unroll
    for (int r = 0; r < 8; ++r) {
        const int row = mBase + r + half * 8;
        const int col = nBase + l16;
        C[((size_t)row * LLAB + l) * DIM + col] = c[r];
    }
}

// ---------------------------------------------------------------------------
// Gather-accumulate: out[row,:] = base[row,:] + sum_i proj[i, labels[row,i], :]
// Block = 256 threads = 8 wave-sized subgroups; subgroup owns one row, each
// lane owns 4 columns (float4 loads: 512B per VMEM instr per wave against the
// L2-resident proj buffer). Label rows staged in 32KB LDS.
// Fuses new = base + msg and emits the bf16 copy for the next projection.
// ---------------------------------------------------------------------------
__global__ __launch_bounds__(256)
void gather_rows(const float* __restrict__ base,
                 const int*   __restrict__ labels,  // [1024, 1024] row-major
                 const float* __restrict__ proj,    // [1024, L, 128]
                 float* __restrict__ outF,
                 bf16*  __restrict__ outB)
{
    __shared__ int lab[8][NAGT];
    const int sub  = threadIdx.x >> 5;    // 0..7: row within block
    const int lane = threadIdx.x & 31;    // column group: cols lane*4 .. +3
    const int row  = blockIdx.x * 8 + sub;

    // Cooperative, coalesced label staging: 8 rows x 1024 ints.
#pragma unroll
    for (int i = 0; i < 8 * NAGT; i += 256) {
        const int idx = i + threadIdx.x;
        const int r = idx >> 10, c = idx & (NAGT - 1);
        lab[r][c] = labels[(size_t)(blockIdx.x * 8 + r) * NAGT + c];
    }
    __syncthreads();

    const int c0 = lane * 4;
    f32x4 acc = *(const f32x4*)(base + (size_t)row * DIM + c0);
#pragma unroll 8
    for (int t = 0; t < NAGT; ++t) {
        const int l = lab[sub][t];                       // LDS broadcast
        acc += *(const f32x4*)(proj + ((size_t)t * LLAB + l) * DIM + c0);
    }
    *(f32x4*)(outF + (size_t)row * DIM + c0) = acc;

    bf16x4 accb;
#pragma unroll
    for (int i = 0; i < 4; ++i) accb[i] = (bf16)acc[i];
    *(bf16x4*)(outB + (size_t)row * DIM + c0) = accb;    // one 8B store
}

// ---------------------------------------------------------------------------
// Weight transpose + cast:  dst[l][cidx][r] = (bf16) src[l][r][cidx]
// (so the WMMA B operand sees contiguous K)
// ---------------------------------------------------------------------------
__global__ __launch_bounds__(128)
void wcast_t(const float* __restrict__ src, bf16* __restrict__ dst)
{
    const int l    = blockIdx.y;
    const int cidx = blockIdx.x;     // column in src, row in dst
    const int r    = threadIdx.x;    // row in src, contiguous in dst
    dst[((size_t)l * DIM + cidx) * DIM + r] =
        (bf16)src[((size_t)l * DIM + r) * DIM + cidx];
}

// ---------------------------------------------------------------------------
// Tiled int32 transpose (inputs -> inputsT) so the T-phase gather's label
// loads are coalesced.
// ---------------------------------------------------------------------------
__global__ __launch_bounds__(1024)
void transpose_i32(const int* __restrict__ src, int* __restrict__ dst)
{
    __shared__ int tile[32][33];
    const int x = blockIdx.x * 32 + threadIdx.x;
    const int y = blockIdx.y * 32 + threadIdx.y;
    tile[threadIdx.y][threadIdx.x] = src[(size_t)y * NAGT + x];
    __syncthreads();
    const int tx = blockIdx.y * 32 + threadIdx.x;
    const int ty = blockIdx.x * 32 + threadIdx.y;
    dst[(size_t)ty * NAGT + tx] = tile[threadIdx.x][threadIdx.y];
}

__global__ __launch_bounds__(256)
void cast_bf(const float* __restrict__ src, bf16* __restrict__ dst, int n)
{
    const int i = blockIdx.x * blockDim.x + threadIdx.x;
    if (i < n) dst[i] = (bf16)src[i];
}

// Output: [2048, 256]; top = [a | 0], bottom = [t | 0]
__global__ __launch_bounds__(256)
void assemble_out(const float* __restrict__ aF, const float* __restrict__ tF,
                  float* __restrict__ out)
{
    const int r = blockIdx.x;
    const int c = threadIdx.x;
    float v = 0.0f;
    if (c < DIM)
        v = (r < NAGT) ? aF[(size_t)r * DIM + c]
                       : tF[(size_t)(r - NAGT) * DIM + c];
    out[(size_t)r * 256 + c] = v;
}

// ---------------------------------------------------------------------------
extern "C" void kernel_launch(void* const* d_in, const int* in_sizes, int n_in,
                              void* d_out, int out_size, void* d_ws, size_t ws_size,
                              hipStream_t stream)
{
    (void)in_sizes; (void)n_in; (void)out_size; (void)ws_size;

    const int*   inputs  = (const int*)  d_in[0];  // [1024,1024]
    const float* first_a = (const float*)d_in[1];  // [1024,128]
    const float* first_t = (const float*)d_in[2];  // [1024,128]
    const float* Awij    = (const float*)d_in[3];  // [L,A,E]
    const float* Awij2   = (const float*)d_in[4];  // [L,E,A]
    float*       out     = (float*)d_out;

    // Workspace carve-up (256B aligned)
    char* ws = (char*)d_ws;
    size_t off = 0;
    auto carve = [&](size_t bytes) -> char* {
        char* p = ws + off;
        off += (bytes + 255) & ~(size_t)255;
        return p;
    };
    float* proj  = (float*)carve((size_t)NAGT * LLAB * DIM * 4); // 64 MB
    float* aF    = (float*)carve((size_t)NAGT * DIM * 4);
    float* tF    = (float*)carve((size_t)NAGT * DIM * 4);
    bf16*  aB    = (bf16*) carve((size_t)NAGT * DIM * 2);
    bf16*  tB    = (bf16*) carve((size_t)NAGT * DIM * 2);
    bf16*  W1bt  = (bf16*) carve((size_t)LLAB * DIM * DIM * 2);  // Awij^T  [l][e][a]
    bf16*  W2bt  = (bf16*) carve((size_t)LLAB * DIM * DIM * 2);  // Awij2^T [l][a][e]
    int*   inT   = (int*)  carve((size_t)NAGT * NAGT * 4);       // inputs^T

    // One-time prep
    wcast_t<<<dim3(DIM, LLAB), 128, 0, stream>>>(Awij,  W1bt);
    wcast_t<<<dim3(DIM, LLAB), 128, 0, stream>>>(Awij2, W2bt);
    transpose_i32<<<dim3(32, 32), dim3(32, 32), 0, stream>>>(inputs, inT);
    cast_bf<<<(NAGT * DIM + 255) / 256, 256, 0, stream>>>(first_t, tB, NAGT * DIM);

    const dim3 projGrid(NAGT / 16, LLAB);

    for (int step = 0; step < 2; ++step) {
        const float* aSrc = step ? aF : first_a;
        const float* tSrc = step ? tF : first_t;

        // Phase A: proj[t,l,:] = t_state @ Awij2[l];  a += gather
        proj_wmma<<<projGrid, 256, 0, stream>>>(tB, W2bt, proj);
        gather_rows<<<NAGT / 8, 256, 0, stream>>>(aSrc, inputs, proj, aF, aB);

        // Phase T: proj[j,l,:] = new_a @ Awij[l];     t += gather
        proj_wmma<<<projGrid, 256, 0, stream>>>(aB, W1bt, proj);
        gather_rows<<<NAGT / 8, 256, 0, stream>>>(tSrc, inT, proj, tF, tB);
    }

    assemble_out<<<2 * NAGT, 256, 0, stream>>>(aF, tF, out);
}
